// AttentionCodeVectorizer_40063454937143
// MI455X (gfx1250) — matrix operand order repaired and checked by reading, not compiled, not run on hardware
//
#include <hip/hip_runtime.h>
#include <math.h>

typedef __attribute__((ext_vector_type(16))) __bf16 v16bf;
typedef __attribute__((ext_vector_type(8)))  float  v8f;

#define N_CTX      200
#define NPAD       224          // padded rows: 14 m-tiles of 16
#define D_DIM      128
#define F_DIM      384
#define KT_N       12           // 384 / 32
#define A_STRIDE   40           // ushorts per row of A/W tiles (32 + 8 pad, 80B)
#define COMB_STR   129          // floats per comb row (128 + 1 pad)

// ---- dynamic LDS layout (bytes, all 16B aligned) ----
#define COMB_OFF   0
#define COMB_BYTES (NPAD * COMB_STR * 4)          // 115584
#define A_OFF      (COMB_OFF + COMB_BYTES)        // 115584
#define A_BYTES    (NPAD * A_STRIDE * 2)          // 17920
#define W_OFF      (A_OFF + A_BYTES)              // 133504
#define W_BYTES    (128 * A_STRIDE * 2)           // 10240
#define IDX_OFF    (W_OFF + W_BYTES)              // 143744
#define IDX_BYTES  (3 * NPAD * 4)                 // 2688
#define ATTV_OFF   (IDX_OFF + IDX_BYTES)          // 146432
#define SC_OFF     (ATTV_OFF + 512)               // 146944
#define RED_OFF    (SC_OFF + NPAD * 4)            // 147840
#define CODE_OFF   (RED_OFF + 16)                 // 147856
#define SMEM_BYTES (CODE_OFF + 512)               // 148368  (~145 KB)

__device__ __forceinline__ unsigned short f2bf(float f) {
    unsigned int u = __float_as_uint(f);
    u += 0x7fffu + ((u >> 16) & 1u);             // round to nearest even
    return (unsigned short)(u >> 16);
}

union Frag { uint4 u[2]; v16bf v; };

__global__ void convert_w_bf16(const float* __restrict__ W,
                               unsigned short* __restrict__ Wbf, int n) {
    int i = blockIdx.x * blockDim.x + threadIdx.x;
    if (i < n) Wbf[i] = f2bf(W[i]);
}

__global__ __launch_bounds__(512)
void acv_fused(const int* __restrict__ xs_xt,        // [B, N, 2]
               const int* __restrict__ path_idx,     // [B, N]
               const float* __restrict__ value_vocab,// [100000, 128]
               const float* __restrict__ path_vocab, // [500000, 128]
               const unsigned short* __restrict__ Wbf,// [128, 384] bf16
               const float* __restrict__ attv,       // [128]
               const float* __restrict__ dense_w,    // [128, 128]
               const float* __restrict__ dense_b,    // [128]
               float* __restrict__ out)              // [B, 128]
{
    extern __shared__ __align__(16) char smem[];
    float*          comb   = (float*)(smem + COMB_OFF);
    unsigned short* Alds   = (unsigned short*)(smem + A_OFF);
    unsigned short* Wlds   = (unsigned short*)(smem + W_OFF);
    int*            idxl   = (int*)(smem + IDX_OFF);
    float*          attv_l = (float*)(smem + ATTV_OFF);
    float*          sc     = (float*)(smem + SC_OFF);
    float*          red    = (float*)(smem + RED_OFF);
    float*          codev  = (float*)(smem + CODE_OFF);

    const int b    = blockIdx.x;
    const int tid  = threadIdx.x;
    const int lane = tid & 31;
    const int wid  = tid >> 5;
    const int nt     = wid & 7;         // n-tile (output column tile)
    const int mbase  = (wid >> 3) * 7;  // 7 m-tiles per wave-group

    // ---- preload indices + attention vector ----
    if (tid < NPAD) {
        int xs = 0, pj = 0, xt = 0;
        if (tid < N_CTX) {
            xs = xs_xt[(b * N_CTX + tid) * 2];
            xt = xs_xt[(b * N_CTX + tid) * 2 + 1];
            pj = path_idx[b * N_CTX + tid];
        }
        idxl[0 * NPAD + tid] = xs;
        idxl[1 * NPAD + tid] = pj;
        idxl[2 * NPAD + tid] = xt;
    }
    if (tid < 128) attv_l[tid] = attv[tid];

    const int row16  = lane & 15;
    const int akb    = (lane >> 4) << 3;   // A frag: 0 or 8 (split-K interleave)
    const int bks    = (lane >> 4) << 3;   // B frag: 0 or 8 (split-K interleave)

    v8f acc[7];
#pragma unroll
    for (int mi = 0; mi < 7; ++mi) acc[mi] = (v8f){0.f,0.f,0.f,0.f,0.f,0.f,0.f,0.f};

    // ================= GEMM: comb = tanh(ctx @ W^T) =================
    for (int kt = 0; kt < KT_N; ++kt) {
        __syncthreads();   // previous iteration's fragment reads done

        // segment: kt 0-3 -> xs(value), 4-7 -> path, 8-11 -> xt(value)
        const int seg = kt >> 2;
        const int off = (kt & 3) * 32;
        const float* table = (seg == 1) ? path_vocab : value_vocab;
        const int*   idxp  = idxl + seg * NPAD;

        // gather A tile [NPAD x 32] f32 -> bf16 in LDS (1792 float4-tasks)
        for (int task = tid; task < NPAD * 8; task += 512) {
            const int row = task >> 3;
            const int c4  = (task & 7) * 4;
            float4 v = make_float4(0.f, 0.f, 0.f, 0.f);
            if (row < N_CTX) {
                const float* src = table + (long)idxp[row] * 128 + off + c4;
                v = *reinterpret_cast<const float4*>(src);
            }
            ushort4 o;
            o.x = f2bf(v.x); o.y = f2bf(v.y); o.z = f2bf(v.z); o.w = f2bf(v.w);
            *reinterpret_cast<ushort4*>(Alds + row * A_STRIDE + c4) = o;
        }
        // load W tile [128 x 32] bf16 (512 tasks of 8 bf16)
        {
            const int r  = tid >> 2;
            const int s8 = (tid & 3) * 8;
            const uint4 w4 = *reinterpret_cast<const uint4*>(
                Wbf + r * F_DIM + kt * 32 + s8);
            *reinterpret_cast<uint4*>(Wlds + r * A_STRIDE + s8) = w4;
        }
        __syncthreads();

        // B fragment: column n = nt*16 + row16; split-K interleave like A
        Frag fb;
        {
            const unsigned short* wb = Wlds + (nt * 16 + row16) * A_STRIDE + bks;
            fb.u[0] = *reinterpret_cast<const uint4*>(wb);
            fb.u[1] = *reinterpret_cast<const uint4*>(wb + 16);
        }
#pragma unroll
        for (int mi = 0; mi < 7; ++mi) {
            const unsigned short* ab =
                Alds + ((mbase + mi) * 16 + row16) * A_STRIDE + akb;
            Frag fa;
            fa.u[0] = *reinterpret_cast<const uint4*>(ab);        // K kb..kb+7
            fa.u[1] = *reinterpret_cast<const uint4*>(ab + 16);   // K 16+kb..
            acc[mi] = __builtin_amdgcn_wmma_f32_16x16x32_bf16(
                false, fa.v, false, fb.v, (short)0, acc[mi], false, false);
        }
    }

    // ---- tanh + store comb to LDS ----
    {
        const int ccol = nt * 16 + row16;
        const int rsub = (lane >> 4) << 3;   // lanes 16-31 hold M = v+8
#pragma unroll
        for (int mi = 0; mi < 7; ++mi) {
            const int rbase = (mbase + mi) * 16 + rsub;
#pragma unroll
            for (int v = 0; v < 8; ++v)
                comb[(rbase + v) * COMB_STR + ccol] = tanhf(acc[mi][v]);
        }
    }
    __syncthreads();

    // ================= attention softmax =================
    if (tid < NPAD) {
        float s = -INFINITY;
        if (tid < N_CTX) {
            s = 0.f;
            const float* cr = comb + tid * COMB_STR;   // stride 129: no bank conflicts
            for (int d = 0; d < D_DIM; ++d) s += cr[d] * attv_l[d];
        }
        sc[tid] = s;
    }
    __syncthreads();
    if (tid < 32) {
        float m = -INFINITY;
        for (int i = tid; i < NPAD; i += 32) m = fmaxf(m, sc[i]);
        for (int o = 16; o; o >>= 1) m = fmaxf(m, __shfl_xor(m, o, 32));
        if (tid == 0) red[0] = m;
    }
    __syncthreads();
    {
        const float mx = red[0];
        if (tid < NPAD) sc[tid] = (tid < N_CTX) ? __expf(sc[tid] - mx) : 0.f;
    }
    __syncthreads();
    if (tid < 32) {
        float s = 0.f;
        for (int i = tid; i < NPAD; i += 32) s += sc[i];
        for (int o = 16; o; o >>= 1) s += __shfl_xor(s, o, 32);
        if (tid == 0) red[1] = 1.0f / s;
    }
    __syncthreads();

    // ---- code = sum_n att[n] * comb[n, :] ----
    if (tid < D_DIM) {
        float a = 0.f;
        for (int n = 0; n < N_CTX; ++n) a += sc[n] * comb[n * COMB_STR + tid];
        codev[tid] = a * red[1];
    }
    __syncthreads();

    // ---- out = sigmoid(code @ dense_w + dense_b) ----
    if (tid < D_DIM) {
        float a = dense_b[tid];
        for (int d = 0; d < D_DIM; ++d) a += codev[d] * dense_w[d * D_DIM + tid];
        out[b * D_DIM + tid] = 1.0f / (1.0f + __expf(-a));
    }
}

extern "C" void kernel_launch(void* const* d_in, const int* in_sizes, int n_in,
                              void* d_out, int out_size, void* d_ws, size_t ws_size,
                              hipStream_t stream) {
    const int*   xs_xt       = (const int*)d_in[0];
    const int*   path_idx    = (const int*)d_in[1];
    const float* value_vocab = (const float*)d_in[2];
    const float* path_vocab  = (const float*)d_in[3];
    const float* W           = (const float*)d_in[4];
    const float* attv        = (const float*)d_in[5];
    const float* dense_w     = (const float*)d_in[6];
    const float* dense_b     = (const float*)d_in[7];
    float*       out         = (float*)d_out;

    unsigned short* Wbf = (unsigned short*)d_ws;   // 128*384 bf16 = 96 KB
    const int wn = 128 * 384;
    convert_w_bf16<<<(wn + 255) / 256, 256, 0, stream>>>(W, Wbf, wn);

    acv_fused<<<1024, 512, SMEM_BYTES, stream>>>(
        xs_xt, path_idx, value_vocab, path_vocab, Wbf,
        attv, dense_w, dense_b, out);
}
